// MambaModel_30502857736726
// MI455X (gfx1250) — matrix operand order repaired and checked by reading, not compiled, not run on hardware
//
#include <hip/hip_runtime.h>
#include <hip/hip_bf16.h>
#include <cstdint>
#include <cstddef>

#define NL   4
#define DM   1024
#define DI   2048
#define DS_  16
#define DC   4
#define DTR  64
#define IN_  256
#define OUT_ 128
#define BATCH 4096

// ---------------- WMMA types (CDNA5 wave32) ----------------
typedef __bf16 v16bf __attribute__((ext_vector_type(16)));
typedef float  v8f   __attribute__((ext_vector_type(8)));

union FragA { v16bf v; unsigned int u[8]; };
union FragC { v8f  v; float f[8]; };

static __device__ __forceinline__ unsigned short f2bf(float f) {
  unsigned int u = __float_as_uint(f);
  u += 0x7FFFu + ((u >> 16) & 1u);   // round-to-nearest-even
  return (unsigned short)(u >> 16);
}
static __device__ __forceinline__ float bf2f(unsigned short h) {
  return __uint_as_float(((unsigned int)h) << 16);
}
static __device__ __forceinline__ float sigm(float x) { return 1.f / (1.f + __expf(-x)); }

// ---------------- bf16 WMMA GEMM: C[M,N] = A[M,K] * W[N,K]^T (+bias) ----------------
// Tiles stream global -> LDS via GLOBAL_LOAD_ASYNC_TO_LDS_B128 (ASYNCcnt),
// double-buffered so tile t+1 loads behind tile t's WMMAs.
#define BM 128
#define BN 128
#define BK 32
#define LDT 40   // halfwords per LDS row: 80B stride, 16B aligned for async b128

__global__ __launch_bounds__(256)
void k_gemm(const unsigned short* __restrict__ A,
            const unsigned short* __restrict__ W,
            float* __restrict__ C,
            const float* __restrict__ bias,
            int N, int K)
{
  __shared__ unsigned short As[2][BM * LDT];
  __shared__ unsigned short Bs[2][BM * LDT];

  const int t    = threadIdx.x;
  const int lane = t & 31;
  const int wave = t >> 5;
  const int wm   = wave >> 1;     // 0..3 : 32-row slab
  const int wn   = wave & 1;      // 0..1 : 64-col slab
  const int half = lane >> 4;
  const int l16  = lane & 15;
  const int bm0  = blockIdx.y * BM;
  const int bn0  = blockIdx.x * BN;

  const int r0 = t >> 2;          // 0..63 ; second chunk is r0+64
  const int s0 = t & 3;           // 16B chunk within the 64B row

  FragC acc[2][4];
  #pragma unroll
  for (int i = 0; i < 2; ++i)
    #pragma unroll
    for (int j = 0; j < 4; ++j)
      #pragma unroll
      for (int e = 0; e < 8; ++e) acc[i][j].f[e] = 0.f;

  // async global->LDS: each wave issues exactly 2 (A) + 2 (B) b128 copies per tile
  auto issueA = [&](int buf, int k0) {
    #pragma unroll
    for (int i = 0; i < 2; ++i) {
      int r = r0 + i * 64;
      unsigned lds = (unsigned)(size_t)(&As[buf][r * LDT + s0 * 8]);
      const unsigned short* ga = A + (size_t)(bm0 + r) * K + k0 + s0 * 8;
      asm volatile("global_load_async_to_lds_b128 %0, %1, off"
                   :: "v"(lds), "v"(ga) : "memory");
    }
  };
  auto issueB = [&](int buf, int k0, bool checked) {
    #pragma unroll
    for (int i = 0; i < 2; ++i) {
      int r = r0 + i * 64;
      int n = bn0 + r;
      unsigned lds = (unsigned)(size_t)(&Bs[buf][r * LDT + s0 * 8]);
      const unsigned short* ga = W + (size_t)n * K + k0 + s0 * 8;
      if (!checked || n < N)
        asm volatile("global_load_async_to_lds_b128 %0, %1, off"
                     :: "v"(lds), "v"(ga) : "memory");
    }
  };

  auto mma = [&](int buf) {
    FragA af[2], bfv[4];
    // A per ISA: lanes 0-15 M=0..15; vgpr j holds pair k=(j&3)*2 + half*8 + (j&4?16:0)
    #pragma unroll
    for (int i = 0; i < 2; ++i) {
      const unsigned short* ap = &As[buf][(wm * 32 + i * 16 + l16) * LDT];
      #pragma unroll
      for (int j = 0; j < 8; ++j) {
        int kk = ((j & 3) * 2) + half * 8 + ((j & 4) ? 16 : 0);
        af[i].u[j] = *(const unsigned int*)(ap + kk);
      }
    }
    // B: lane = N col; vgpr j pair k=(j&3)*2 + half*8 + (j&4?16:0) (same as A)
    #pragma unroll
    for (int i = 0; i < 4; ++i) {
      const unsigned short* bp = &Bs[buf][(wn * 64 + i * 16 + l16) * LDT];
      #pragma unroll
      for (int j = 0; j < 8; ++j)
        bfv[i].u[j] = *(const unsigned int*)(bp + ((j & 3) * 2) + half * 8 + ((j & 4) ? 16 : 0));
    }
    #pragma unroll
    for (int i = 0; i < 2; ++i)
      #pragma unroll
      for (int j = 0; j < 4; ++j)
        acc[i][j].v = __builtin_amdgcn_wmma_f32_16x16x32_bf16(
            false, af[i].v, false, bfv[j].v, (short)0, acc[i][j].v, false, false);
  };

  if (bn0 + BN <= N) {
    // -------- full-N fast path: double-buffered async pipeline --------
    issueA(0, 0);
    issueB(0, 0, false);
    const int nt = K / BK;
    for (int ti = 0; ti < nt; ++ti) {
      const int buf = ti & 1;
      if (ti + 1 < nt) {
        issueA(buf ^ 1, (ti + 1) * BK);
        issueB(buf ^ 1, (ti + 1) * BK, false);
        // retire current tile's 4 async ops (in-order); next tile keeps streaming
        asm volatile("s_wait_asynccnt 0x4" ::: "memory");
      } else {
        asm volatile("s_wait_asynccnt 0x0" ::: "memory");
      }
      __syncthreads();
      mma(buf);
      __syncthreads();
    }
  } else {
    // -------- partial-N path (x_proj, N=96): single buffer, masked B --------
    const int nvalid = N - bn0;
    for (int idx = t; idx < (BM - nvalid) * 16; idx += 256) {
      int r = nvalid + (idx >> 4);
      ((unsigned int*)&Bs[0][r * LDT])[idx & 15] = 0u;   // OOB rows stay zero
    }
    __syncthreads();
    for (int k0 = 0; k0 < K; k0 += BK) {
      issueA(0, k0);
      issueB(0, k0, true);
      asm volatile("s_wait_asynccnt 0x0" ::: "memory");
      __syncthreads();
      mma(0);
      __syncthreads();
    }
  }

  // D layout: vgpr e -> m = e + 8*half ; n = lane&15
  #pragma unroll
  for (int i = 0; i < 2; ++i) {
    int mbase = bm0 + wm * 32 + i * 16 + half * 8;
    #pragma unroll
    for (int j = 0; j < 4; ++j) {
      int n = bn0 + wn * 64 + j * 16 + l16;
      if (n < N) {
        float bv = bias ? bias[n] : 0.f;
        #pragma unroll
        for (int e = 0; e < 8; ++e)
          C[(size_t)(mbase + e) * N + n] = acc[i][j].f[e] + bv;
      }
    }
  }
}

// ---------------- elementwise / glue kernels ----------------
__global__ void k_cvt(const float* __restrict__ in, unsigned short* __restrict__ out, int n) {
  int i = blockIdx.x * blockDim.x + threadIdx.x;
  if (i < n) out[i] = f2bf(in[i]);
}

// xc = silu(conv_b + xz[:, :DI] * conv_w[:, DC-1])   (L==1: only last tap survives)
__global__ void k_conv_silu(const float* __restrict__ xz, const float* __restrict__ cw,
                            const float* __restrict__ cb, unsigned short* __restrict__ xc_bf) {
  int i = blockIdx.x * blockDim.x + threadIdx.x;
  if (i >= BATCH * DI) return;
  int b = i / DI, d = i - b * DI;
  float v = cb[d] + xz[(size_t)b * (2 * DI) + d] * cw[d * DC + (DC - 1)];
  xc_bf[i] = f2bf(v * sigm(v));
}

// dt_bf = bf16(x_dbl[:, :64]); bc[b] = sum_n B[b,n]*C[b,n]
__global__ void k_xdbl_post(const float* __restrict__ xd, unsigned short* __restrict__ dt_bf,
                            float* __restrict__ bc) {
  int b = blockIdx.x, t = threadIdx.x;           // blockDim = 64
  const float* row = xd + (size_t)b * (DTR + 2 * DS_);
  dt_bf[(size_t)b * DTR + t] = f2bf(row[t]);
  if (t < DS_) {
    float p = row[DTR + t] * row[DTR + DS_ + t];
    #pragma unroll
    for (int off = 8; off > 0; off >>= 1) p += __shfl_down(p, off, 16);
    if (t == 0) bc[b] = p;
  }
}

// dt = softplus(dtf + dt_b); y = (dt*bc[b] + Dp)*xc; y *= silu(z)
__global__ void k_scan_gate(const float* __restrict__ dtf, const float* __restrict__ dt_b,
                            const unsigned short* __restrict__ xc_bf, const float* __restrict__ bc,
                            const float* __restrict__ Dp, const float* __restrict__ xz,
                            unsigned short* __restrict__ y_bf) {
  int i = blockIdx.x * blockDim.x + threadIdx.x;
  if (i >= BATCH * DI) return;
  int b = i / DI, d = i - b * DI;
  float s = dtf[i] + dt_b[d];
  float dt = (s > 20.f) ? s : log1pf(expf(s));
  float xc = bf2f(xc_bf[i]);
  float y = dt * xc * bc[b] + Dp[d] * xc;
  float z = xz[(size_t)b * (2 * DI) + DI + d];
  y *= z * sigm(z);
  y_bf[i] = f2bf(y);
}

// h[b,:] += layer_norm(mb[b,:]) * g + beta   (DM = 1024, 256 threads x 4)
__global__ __launch_bounds__(256)
void k_ln_res(const float* __restrict__ mb, const float* __restrict__ g,
              const float* __restrict__ bta, float* __restrict__ h) {
  __shared__ float red[8];
  int b = blockIdx.x, t = threadIdx.x;
  int lane = t & 31, wave = t >> 5;
  const float* row = mb + (size_t)b * DM;
  float v[4]; float s = 0.f;
  #pragma unroll
  for (int i = 0; i < 4; ++i) { v[i] = row[t + i * 256]; s += v[i]; }
  #pragma unroll
  for (int off = 16; off > 0; off >>= 1) s += __shfl_xor(s, off, 32);
  if (lane == 0) red[wave] = s;
  __syncthreads();
  float mu = 0.f;
  #pragma unroll
  for (int w = 0; w < 8; ++w) mu += red[w];
  mu *= (1.f / DM);
  __syncthreads();
  float vs = 0.f;
  #pragma unroll
  for (int i = 0; i < 4; ++i) { float dd = v[i] - mu; vs += dd * dd; }
  #pragma unroll
  for (int off = 16; off > 0; off >>= 1) vs += __shfl_xor(vs, off, 32);
  if (lane == 0) red[wave] = vs;
  __syncthreads();
  float var = 0.f;
  #pragma unroll
  for (int w = 0; w < 8; ++w) var += red[w];
  float rstd = rsqrtf(var * (1.f / DM) + 1e-5f);
  #pragma unroll
  for (int i = 0; i < 4; ++i) {
    int d = t + i * 256;
    h[(size_t)b * DM + d] += (v[i] - mu) * rstd * g[d] + bta[d];
  }
}

// ---------------- orchestration ----------------
extern "C" void kernel_launch(void* const* d_in, const int* in_sizes, int n_in,
                              void* d_out, int out_size, void* d_ws, size_t ws_size,
                              hipStream_t stream) {
  (void)in_sizes; (void)n_in; (void)out_size; (void)ws_size;
  const float* x      = (const float*)d_in[0];
  const float* inp_w  = (const float*)d_in[1];
  const float* inp_b  = (const float*)d_in[2];
  const float* ipw    = (const float*)d_in[3];
  const float* conv_w = (const float*)d_in[4];
  const float* conv_b = (const float*)d_in[5];
  const float* xp_w   = (const float*)d_in[6];
  const float* dt_w   = (const float*)d_in[7];
  const float* dt_b   = (const float*)d_in[8];
  /* d_in[9] = A_log : unused — L==1 and h0==0 eliminate exp(dt*A)*h */
  const float* Dp     = (const float*)d_in[10];
  const float* out_w  = (const float*)d_in[11];
  const float* ln_g   = (const float*)d_in[12];
  const float* ln_b   = (const float*)d_in[13];
  const float* outp_w = (const float*)d_in[14];
  const float* outp_b = (const float*)d_in[15];
  float* out = (float*)d_out;

  char* p = (char*)d_ws;
  auto alloc = [&](size_t bytes) { char* r = p; p += (bytes + 255) & ~(size_t)255; return r; };

  unsigned short* x_bf    = (unsigned short*)alloc((size_t)BATCH * IN_ * 2);
  unsigned short* inpw_bf = (unsigned short*)alloc((size_t)DM * IN_ * 2);
  unsigned short* ipw_bf  = (unsigned short*)alloc((size_t)NL * 2 * DI * DM * 2);
  unsigned short* xpw_bf  = (unsigned short*)alloc((size_t)NL * (DTR + 2 * DS_) * DI * 2);
  unsigned short* dtw_bf  = (unsigned short*)alloc((size_t)NL * DI * DTR * 2);
  unsigned short* outw_bf = (unsigned short*)alloc((size_t)NL * DM * DI * 2);
  unsigned short* opw_bf  = (unsigned short*)alloc((size_t)OUT_ * DM * 2);
  float*          h       = (float*)alloc((size_t)BATCH * DM * 4);
  unsigned short* h_bf    = (unsigned short*)alloc((size_t)BATCH * DM * 2);
  float*          xz      = (float*)alloc((size_t)BATCH * 2 * DI * 4);
  unsigned short* xc_bf   = (unsigned short*)alloc((size_t)BATCH * DI * 2);
  float*          x_dbl   = (float*)alloc((size_t)BATCH * (DTR + 2 * DS_) * 4);
  unsigned short* dt_bf   = (unsigned short*)alloc((size_t)BATCH * DTR * 2);
  float*          bc      = (float*)alloc((size_t)BATCH * 4);
  float*          dtf     = (float*)alloc((size_t)BATCH * DI * 4);
  unsigned short* y_bf    = (unsigned short*)alloc((size_t)BATCH * DI * 2);
  float*          mb      = dtf;  // dtf fully consumed before out-proj writes here

  auto cvt = [&](const float* src, unsigned short* dst, size_t n) {
    k_cvt<<<dim3((unsigned)((n + 255) / 256)), dim3(256), 0, stream>>>(src, dst, (int)n);
  };
  cvt(x, x_bf, (size_t)BATCH * IN_);
  cvt(inp_w, inpw_bf, (size_t)DM * IN_);
  cvt(ipw, ipw_bf, (size_t)NL * 2 * DI * DM);
  cvt(xp_w, xpw_bf, (size_t)NL * (DTR + 2 * DS_) * DI);
  cvt(dt_w, dtw_bf, (size_t)NL * DI * DTR);
  cvt(out_w, outw_bf, (size_t)NL * DM * DI);
  cvt(outp_w, opw_bf, (size_t)OUT_ * DM);

  auto gemm = [&](const unsigned short* A, const unsigned short* W, float* C,
                  const float* bias, int N, int K) {
    dim3 grid((unsigned)((N + BN - 1) / BN), BATCH / BM);
    k_gemm<<<grid, dim3(256), 0, stream>>>(A, W, C, bias, N, K);
  };

  gemm(x_bf, inpw_bf, h, inp_b, DM, IN_);

  const unsigned EWG = (unsigned)(((size_t)BATCH * DI + 255) / 256);
  for (int i = 0; i < NL; ++i) {
    cvt(h, h_bf, (size_t)BATCH * DM);
    gemm(h_bf, ipw_bf + (size_t)i * 2 * DI * DM, xz, nullptr, 2 * DI, DM);
    k_conv_silu<<<EWG, 256, 0, stream>>>(xz, conv_w + (size_t)i * DI * DC,
                                         conv_b + (size_t)i * DI, xc_bf);
    gemm(xc_bf, xpw_bf + (size_t)i * (DTR + 2 * DS_) * DI, x_dbl, nullptr, DTR + 2 * DS_, DI);
    k_xdbl_post<<<BATCH, 64, 0, stream>>>(x_dbl, dt_bf, bc);
    gemm(dt_bf, dtw_bf + (size_t)i * DI * DTR, dtf, nullptr, DI, DTR);
    k_scan_gate<<<EWG, 256, 0, stream>>>(dtf, dt_b + (size_t)i * DI, xc_bf, bc,
                                         Dp + (size_t)i * DI, xz, y_bf);
    gemm(y_bf, outw_bf + (size_t)i * DM * DI, mb, nullptr, DM, DI);
    k_ln_res<<<BATCH, 256, 0, stream>>>(mb, ln_g + (size_t)i * DM, ln_b + (size_t)i * DM, h);
  }
  cvt(h, h_bf, (size_t)BATCH * DM);
  gemm(h_bf, opw_bf, out, outp_b, OUT_, DM);
}
